// EnergyTransformerEncoderLayer_51402168598868
// MI455X (gfx1250) — matrix operand order repaired and checked by reading, not compiled, not run on hardware
//
#include <hip/hip_runtime.h>
#include <cstdint>
#include <cstddef>

// ---------------------------------------------------------------------------
// Types
// ---------------------------------------------------------------------------
typedef __attribute__((ext_vector_type(16))) __bf16 v16bf;
typedef __attribute__((ext_vector_type(8)))  float  v8f;
typedef __attribute__((ext_vector_type(4)))  int    v4i;

// AS-qualified pointer types for the async global->LDS builtin:
// param0 is "int4 __device__ *" (AS1), param1 is "int4 __shared__ *" (AS3).
typedef __attribute__((address_space(1))) v4i* gv4ip;
typedef __attribute__((address_space(3))) v4i* lv4ip;

union FragBF { uint4 q[2]; v16bf v; };
union Pack8  { __bf16 h[8]; uint4 q; };
union Pack4  { __bf16 h[4]; uint2 q; };

// ---------------------------------------------------------------------------
// Async global->LDS (CDNA5). Guarded so compile never breaks.
// ---------------------------------------------------------------------------
#define USE_ASYNC_LDS 1
#if USE_ASYNC_LDS && defined(__AMDGCN__) && __has_builtin(__builtin_amdgcn_global_load_async_to_lds_b128)
#define HAVE_ASYNC 1
#else
#define HAVE_ASYNC 0
#endif

__device__ __forceinline__ void async_copy16(const void* g, void* l) {
#if HAVE_ASYNC
  __builtin_amdgcn_global_load_async_to_lds_b128((gv4ip)(void*)g, (lv4ip)l, 0, 0);
#endif
}
__device__ __forceinline__ void async_wait_all() {
#if HAVE_ASYNC
#if __has_builtin(__builtin_amdgcn_s_wait_asynccnt)
  __builtin_amdgcn_s_wait_asynccnt(0);
#else
  asm volatile("s_wait_asynccnt 0" ::: "memory");
#endif
#endif
}

// ---------------------------------------------------------------------------
// fp32 -> bf16 bulk convert
// ---------------------------------------------------------------------------
__global__ __launch_bounds__(256) void cvt_f32_to_bf16_k(const float* __restrict__ in,
                                                         __bf16* __restrict__ out,
                                                         long long n) {
  long long i = ((long long)blockIdx.x * 256 + threadIdx.x) * 4;
  if (i + 3 < n) {
    float4 f = *(const float4*)(in + i);
    Pack4 p;
    p.h[0] = (__bf16)f.x; p.h[1] = (__bf16)f.y;
    p.h[2] = (__bf16)f.z; p.h[3] = (__bf16)f.w;
    *(uint2*)(out + i) = p.q;
  }
}

// ---------------------------------------------------------------------------
// Tiled bf16 WMMA GEMM:  C[M,N] = epilogue( A[M,K] * Bt[N,K]^T * scale + bias )
//   AOPT: 0 = A is bf16, 1 = A is fp32 (convert to bf16 while staging to LDS)
//   OUTOPT: 0 = bf16 row-major, 1 = bf16 transposed per-1024 batch (for vT),
//           2 = fp32 row-major
// 256 threads = 8 waves; 128x128x32 tile; wave grid 4(M) x 2(N);
// per-wave 2x4 tiles of v_wmma_f32_16x16x32_bf16.
// ---------------------------------------------------------------------------
template<int AOPT, int OUTOPT, bool RELU>
__global__ __launch_bounds__(256, 2)
void gemm_k(const void* __restrict__ Aall, const __bf16* __restrict__ Btall,
            void* __restrict__ Call, const float* __restrict__ bias,
            int M, int N, int K, float scale,
            long long sA_, long long sB_, long long sC_)
{
  __shared__ __align__(16) __bf16 sA[128 * 32];   // [row][k], 64B rows
  __shared__ __align__(16) __bf16 sB[128 * 32];   // [n][k],   64B rows

  const int t    = threadIdx.x;
  const int lane = t & 31;
  const int wave = t >> 5;
  const int wm   = wave >> 1;         // 0..3
  const int wn   = wave & 1;          // 0..1
  const int lh   = lane >> 4;         // 0..1
  const int lr   = lane & 15;         // 0..15
  const int z    = blockIdx.z;
  const int m0   = blockIdx.y * 128;
  const int n0   = blockIdx.x * 128;

  const __bf16* A16 = (AOPT == 0) ? ((const __bf16*)Aall + (size_t)z * sA_) : nullptr;
  const float*  A32 = (AOPT == 1) ? ((const float*) Aall + (size_t)z * sA_) : nullptr;
  const __bf16* Bt  = Btall + (size_t)z * sB_;

  const int lrow  = t >> 1;           // 0..127: tile row this thread stages
  const int lhalf = t & 1;            // which 16-element half of the 32-wide k

  v8f acc[2][4] = {};

  for (int k0 = 0; k0 < K; k0 += 32) {
    // ---- stage A tile ----
    if (AOPT == 0) {
      const __bf16* ga = A16 + (size_t)(m0 + lrow) * K + k0 + lhalf * 16;
      __bf16* la = &sA[lrow * 32 + lhalf * 16];
#if HAVE_ASYNC
      async_copy16(ga,     la);
      async_copy16(ga + 8, la + 8);
#else
      if (k0 + 32 < K) __builtin_prefetch(ga + 32, 0, 0);
      uint4 d0 = *(const uint4*)ga;
      uint4 d1 = *(const uint4*)(ga + 8);
      *(uint4*)la       = d0;
      *(uint4*)(la + 8) = d1;
#endif
    } else {
      const float* ga = A32 + (size_t)(m0 + lrow) * K + k0 + lhalf * 16;
      if (k0 + 32 < K) __builtin_prefetch(ga + 32, 0, 0);
      float4 f0 = *(const float4*)(ga + 0);
      float4 f1 = *(const float4*)(ga + 4);
      float4 f2 = *(const float4*)(ga + 8);
      float4 f3 = *(const float4*)(ga + 12);
      Pack8 p0, p1;
      p0.h[0]=(__bf16)f0.x; p0.h[1]=(__bf16)f0.y; p0.h[2]=(__bf16)f0.z; p0.h[3]=(__bf16)f0.w;
      p0.h[4]=(__bf16)f1.x; p0.h[5]=(__bf16)f1.y; p0.h[6]=(__bf16)f1.z; p0.h[7]=(__bf16)f1.w;
      p1.h[0]=(__bf16)f2.x; p1.h[1]=(__bf16)f2.y; p1.h[2]=(__bf16)f2.z; p1.h[3]=(__bf16)f2.w;
      p1.h[4]=(__bf16)f3.x; p1.h[5]=(__bf16)f3.y; p1.h[6]=(__bf16)f3.z; p1.h[7]=(__bf16)f3.w;
      *(uint4*)&sA[lrow * 32 + lhalf * 16]     = p0.q;
      *(uint4*)&sA[lrow * 32 + lhalf * 16 + 8] = p1.q;
    }
    // ---- stage B tile ([n][k]) ----
    {
      const __bf16* gb = Bt + (size_t)(n0 + lrow) * K + k0 + lhalf * 16;
      __bf16* lb = &sB[lrow * 32 + lhalf * 16];
#if HAVE_ASYNC
      async_copy16(gb,     lb);
      async_copy16(gb + 8, lb + 8);
#else
      if (k0 + 32 < K) __builtin_prefetch(gb + 32, 0, 0);
      uint4 d0 = *(const uint4*)gb;
      uint4 d1 = *(const uint4*)(gb + 8);
      *(uint4*)lb       = d0;
      *(uint4*)(lb + 8) = d1;
#endif
    }
    async_wait_all();
    __syncthreads();

    // ---- fragments from LDS per ISA VGPR layouts ----
    FragBF afrag[2], bfrag[4];
#pragma unroll
    for (int mt = 0; mt < 2; mt++) {
      // A 16x32 bf16: lane L elem j -> k = j%8 + 8*(2*(j/8) + L/16)
      const char* p = (const char*)sA + (wm * 32 + mt * 16 + lr) * 64 + lh * 16;
      afrag[mt].q[0] = *(const uint4*)p;
      afrag[mt].q[1] = *(const uint4*)(p + 32);
    }
#pragma unroll
    for (int nt = 0; nt < 4; nt++) {
      // B 32x16 bf16 (stored [n][k]): lane L elem j -> k = j%8 + 8*(2*(j/8) + L/16), n = L%16
      const char* p = (const char*)sB + (wn * 64 + nt * 16 + lr) * 64 + lh * 16;
      bfrag[nt].q[0] = *(const uint4*)p;
      bfrag[nt].q[1] = *(const uint4*)(p + 32);
    }
#pragma unroll
    for (int mt = 0; mt < 2; mt++)
#pragma unroll
      for (int nt = 0; nt < 4; nt++)
        acc[mt][nt] = __builtin_amdgcn_wmma_f32_16x16x32_bf16(
            false, afrag[mt].v, false, bfrag[nt].v, (short)0, acc[mt][nt], false, false);
    __syncthreads();
  }

  // ---- epilogue: C layout: VGPR r -> M = r + 8*(lane/16), N = lane%16 ----
#pragma unroll
  for (int mt = 0; mt < 2; mt++) {
#pragma unroll
    for (int nt = 0; nt < 4; nt++) {
      const int ng = n0 + wn * 64 + nt * 16 + lr;
      const float bv = bias ? bias[ng] : 0.0f;
      const int mb = m0 + wm * 32 + mt * 16 + 8 * lh;
      if (OUTOPT == 2) {
        float* C = (float*)Call + (size_t)z * sC_;
#pragma unroll
        for (int r = 0; r < 8; r++) {
          float v = acc[mt][nt][r] * scale + bv;
          if (RELU) v = fmaxf(v, 0.0f);
          C[(size_t)(mb + r) * N + ng] = v;
        }
      } else if (OUTOPT == 0) {
        __bf16* C = (__bf16*)Call + (size_t)z * sC_;
#pragma unroll
        for (int r = 0; r < 8; r++) {
          float v = acc[mt][nt][r] * scale + bv;
          if (RELU) v = fmaxf(v, 0.0f);
          C[(size_t)(mb + r) * N + ng] = (__bf16)v;
        }
      } else {
        // transposed per-1024-row batch: out[b][n][m%1024], 16B packed store
        Pack8 p;
#pragma unroll
        for (int r = 0; r < 8; r++) p.h[r] = (__bf16)(acc[mt][nt][r] * scale + bv);
        const int bb = mb >> 10, ss = mb & 1023;
        __bf16* CT = (__bf16*)Call;
        *(uint4*)&CT[((size_t)bb * N + ng) * 1024 + ss] = p.q;
      }
    }
  }
}

// ---------------------------------------------------------------------------
// Block reductions (256 threads)
// ---------------------------------------------------------------------------
__device__ __forceinline__ float block_sum(float v, float* red) {
  int t = threadIdx.x;
  red[t] = v; __syncthreads();
#pragma unroll
  for (int s = 128; s > 0; s >>= 1) {
    if (t < s) red[t] += red[t + s];
    __syncthreads();
  }
  float r = red[0]; __syncthreads();
  return r;
}
__device__ __forceinline__ float block_max(float v, float* red) {
  int t = threadIdx.x;
  red[t] = v; __syncthreads();
#pragma unroll
  for (int s = 128; s > 0; s >>= 1) {
    if (t < s) red[t] = fmaxf(red[t], red[t + s]);
    __syncthreads();
  }
  float r = red[0]; __syncthreads();
  return r;
}

// ---------------------------------------------------------------------------
// Row softmax in place (row length 1024, one block per row)
// ---------------------------------------------------------------------------
__global__ __launch_bounds__(256) void softmax_inplace_k(float* __restrict__ buf) {
  __shared__ float red[256];
  float* p = buf + (size_t)blockIdx.x * 1024;
  const int t = threadIdx.x;
  float v0 = p[t], v1 = p[t + 256], v2 = p[t + 512], v3 = p[t + 768];
  float m = block_max(fmaxf(fmaxf(v0, v1), fmaxf(v2, v3)), red);
  float e0 = __expf(v0 - m), e1 = __expf(v1 - m), e2 = __expf(v2 - m), e3 = __expf(v3 - m);
  float s = block_sum(e0 + e1 + e2 + e3, red);
  float inv = 1.0f / s;
  p[t] = e0 * inv; p[t + 256] = e1 * inv; p[t + 512] = e2 * inv; p[t + 768] = e3 * inv;
}

// ---------------------------------------------------------------------------
// Fused: softmax(x x^T scores)[b] row + KL accumulation against aw[b-1] row.
// grid: (1024 rows, 15 batch pairs); never materializes self_attn.
// ---------------------------------------------------------------------------
__global__ __launch_bounds__(256) void kl_k(const float* __restrict__ xscores,
                                            const float* __restrict__ aw,
                                            float* __restrict__ kl) {
  __shared__ float red[256];
  const int b = blockIdx.y + 1;
  const size_t row = ((size_t)b * 1024 + blockIdx.x) * 1024;
  const float* pr = xscores + row;
  const float* pa = aw + row - (size_t)1024 * 1024;   // aw[b-1], same s
  const int t = threadIdx.x;

  float v0 = pr[t], v1 = pr[t + 256], v2 = pr[t + 512], v3 = pr[t + 768];
  float m = block_max(fmaxf(fmaxf(v0, v1), fmaxf(v2, v3)), red);
  float e0 = __expf(v0 - m), e1 = __expf(v1 - m), e2 = __expf(v2 - m), e3 = __expf(v3 - m);
  float s = block_sum(e0 + e1 + e2 + e3, red);
  float inv = 1.0f / s;

  float a0 = pa[t], a1 = pa[t + 256], a2 = pa[t + 512], a3 = pa[t + 768];
  float accv = 0.0f;
  accv += a0 * logf(a0 / (e0 * inv + 1e-6f) + 1e-6f);
  accv += a1 * logf(a1 / (e1 * inv + 1e-6f) + 1e-6f);
  accv += a2 * logf(a2 / (e2 * inv + 1e-6f) + 1e-6f);
  accv += a3 * logf(a3 / (e3 * inv + 1e-6f) + 1e-6f);
  float tot = block_sum(accv, red);
  if (t == 0) atomicAdd(kl, tot * (1.0f / 15.0f));
}

// ---------------------------------------------------------------------------
// Residual + LayerNorm over 1024; optional bf16 mirror for next GEMM.
// ---------------------------------------------------------------------------
__global__ __launch_bounds__(256) void ln_k(const float* __restrict__ in1,
                                            const float* __restrict__ in2,
                                            const float* __restrict__ g,
                                            const float* __restrict__ be,
                                            float* __restrict__ out,
                                            __bf16* __restrict__ outb) {
  __shared__ float red[256];
  const size_t row = (size_t)blockIdx.x * 1024;
  const int t = threadIdx.x;
  float x[4];
#pragma unroll
  for (int i = 0; i < 4; i++) {
    int c = t + 256 * i;
    x[i] = in1[row + c] + in2[row + c];
  }
  float mu = block_sum(x[0] + x[1] + x[2] + x[3], red) * (1.0f / 1024.0f);
  float vs = 0.0f;
#pragma unroll
  for (int i = 0; i < 4; i++) { float d = x[i] - mu; vs += d * d; }
  float var = block_sum(vs, red) * (1.0f / 1024.0f);
  float rstd = rsqrtf(var + 1e-5f);
#pragma unroll
  for (int i = 0; i < 4; i++) {
    int c = t + 256 * i;
    float y = (x[i] - mu) * rstd * g[c] + be[c];
    out[row + c] = y;
    if (outb) outb[row + c] = (__bf16)y;
  }
}

__global__ void zero_one_k(float* p) { *p = 0.0f; }

// ---------------------------------------------------------------------------
// Launch
// ---------------------------------------------------------------------------
extern "C" void kernel_launch(void* const* d_in, const int* in_sizes, int n_in,
                              void* d_out, int out_size, void* d_ws, size_t ws_size,
                              hipStream_t stream) {
  (void)in_sizes; (void)n_in; (void)out_size; (void)ws_size;
  // setup_inputs() dict order:
  const float* x   = (const float*)d_in[0];
  const float* Wq  = (const float*)d_in[1];  const float* bq = (const float*)d_in[2];
  const float* Wk  = (const float*)d_in[3];  const float* bk = (const float*)d_in[4];
  const float* Wv  = (const float*)d_in[5];  const float* bv = (const float*)d_in[6];
  const float* Wo  = (const float*)d_in[7];  const float* bo = (const float*)d_in[8];
  const float* g1  = (const float*)d_in[9];  const float* be1 = (const float*)d_in[10];
  const float* g2  = (const float*)d_in[11]; const float* be2 = (const float*)d_in[12];
  const float* W1  = (const float*)d_in[13]; const float* b1  = (const float*)d_in[14];
  const float* W2  = (const float*)d_in[15]; const float* b2  = (const float*)d_in[16];

  constexpr long long BS  = 16LL * 1024;            // 16384 rows
  constexpr long long SS  = 1024LL * 1024;          // per-batch matrix elems
  const float scl = 0.03125f;                       // 1/sqrt(1024)

  // workspace arena (bytes); ffb overlays aw+xs, ff2 overlays attno
  char* ws = (char*)d_ws;
  const size_t MB = 1u << 20;
  __bf16* wqb   = (__bf16*)(ws + 0   * MB);
  __bf16* wkb   = (__bf16*)(ws + 2   * MB);
  __bf16* wvb   = (__bf16*)(ws + 4   * MB);
  __bf16* wob   = (__bf16*)(ws + 6   * MB);
  __bf16* w1b   = (__bf16*)(ws + 8   * MB);
  __bf16* w2b   = (__bf16*)(ws + 16  * MB);
  __bf16* xb    = (__bf16*)(ws + 24  * MB);
  __bf16* qb    = (__bf16*)(ws + 56  * MB);
  __bf16* kb    = (__bf16*)(ws + 88  * MB);
  __bf16* vtb   = (__bf16*)(ws + 120 * MB);   // [b][d][s]
  float*  aw    = (float*) (ws + 152 * MB);   // scores -> softmax in place
  float*  xs    = (float*) (ws + 216 * MB);   // x x^T scores
  __bf16* attnb = (__bf16*)(ws + 280 * MB);
  float*  attno = (float*) (ws + 312 * MB);
  float*  h     = (float*) (ws + 376 * MB);
  __bf16* hb    = (__bf16*)(ws + 440 * MB);
  __bf16* ffb   = (__bf16*)(ws + 152 * MB);   // reuse aw+xs (dead after KL+attn)
  float*  ff2   = (float*) (ws + 312 * MB);   // reuse attno (dead after LN1)

  dim3 blk(256);

  // fp32 -> bf16 conversions
  cvt_f32_to_bf16_k<<<16384, blk, 0, stream>>>(x,  xb,  BS * 1024);
  cvt_f32_to_bf16_k<<<1024,  blk, 0, stream>>>(Wq, wqb, SS);
  cvt_f32_to_bf16_k<<<1024,  blk, 0, stream>>>(Wk, wkb, SS);
  cvt_f32_to_bf16_k<<<1024,  blk, 0, stream>>>(Wv, wvb, SS);
  cvt_f32_to_bf16_k<<<1024,  blk, 0, stream>>>(Wo, wob, SS);
  cvt_f32_to_bf16_k<<<4096,  blk, 0, stream>>>(W1, w1b, 4 * SS);
  cvt_f32_to_bf16_k<<<4096,  blk, 0, stream>>>(W2, w2b, 4 * SS);

  // projections (M=16384, N=K=1024)
  dim3 g_proj(8, 128, 1);
  gemm_k<0, 0, false><<<g_proj, blk, 0, stream>>>(xb, wqb, qb,  bq, 16384, 1024, 1024, 1.0f, 0, 0, 0);
  gemm_k<0, 0, false><<<g_proj, blk, 0, stream>>>(xb, wkb, kb,  bk, 16384, 1024, 1024, 1.0f, 0, 0, 0);
  gemm_k<0, 1, false><<<g_proj, blk, 0, stream>>>(xb, wvb, vtb, bv, 16384, 1024, 1024, 1.0f, 0, 0, 0);

  // batched scores and x x^T (per batch 1024^3)
  dim3 g_b(8, 8, 16);
  gemm_k<0, 2, false><<<g_b, blk, 0, stream>>>(qb, kb, aw, nullptr, 1024, 1024, 1024, scl,  SS, SS, SS);
  gemm_k<0, 2, false><<<g_b, blk, 0, stream>>>(xb, xb, xs, nullptr, 1024, 1024, 1024, 1.0f, SS, SS, SS);

  // softmax(aw) in place, then fused softmax(xs)+KL
  softmax_inplace_k<<<16384, blk, 0, stream>>>(aw);
  float* klp = (float*)d_out + 16777216;
  zero_one_k<<<1, 1, 0, stream>>>(klp);
  dim3 g_kl(1024, 15, 1);
  kl_k<<<g_kl, blk, 0, stream>>>(xs, aw, klp);

  // attn_out = aw @ v  (A fp32 converted on the fly; B = vT [d][t])
  gemm_k<1, 0, false><<<g_b, blk, 0, stream>>>(aw, vtb, attnb, nullptr, 1024, 1024, 1024, 1.0f, SS, SS, SS);
  // O projection (fp32 out)
  gemm_k<0, 2, false><<<g_proj, blk, 0, stream>>>(attnb, wob, attno, bo, 16384, 1024, 1024, 1.0f, 0, 0, 0);

  // h = LN(x + attn_out), bf16 mirror for FFN
  ln_k<<<16384, blk, 0, stream>>>(x, attno, g1, be1, h, hb);

  // FFN
  dim3 g_f1(32, 128, 1);
  gemm_k<0, 0, true ><<<g_f1, blk, 0, stream>>>(hb,  w1b, ffb, b1, 16384, 4096, 1024, 1.0f, 0, 0, 0);
  dim3 g_f2(8, 128, 1);
  gemm_k<0, 2, false><<<g_f2, blk, 0, stream>>>(ffb, w2b, ff2, b2, 16384, 1024, 4096, 1.0f, 0, 0, 0);

  // out = LN(h + ff)
  ln_k<<<16384, blk, 0, stream>>>(h, ff2, g2, be2, (float*)d_out, nullptr);
}
